// PTDLP_51539607750
// MI455X (gfx1250) — matrix operand order repaired and checked by reading, not compiled, not run on hardware
//
#include <hip/hip_runtime.h>
#include <hip/hip_bf16.h>
#include <stdint.h>

#define NUM_CHAINS 10
#define BATCH      128
#define DIM        512
#define MAX_VAL    128
#define MTOT       (NUM_CHAINS * BATCH)   // 1280 rows

typedef __attribute__((ext_vector_type(16))) _Float16 v16h;
typedef __attribute__((ext_vector_type(8)))  _Float16 v8h;
typedef __attribute__((ext_vector_type(4)))  _Float16 v4h;
typedef __attribute__((ext_vector_type(8)))  float    v8f;
typedef __attribute__((ext_vector_type(4)))  int      v4i;

// Pointer-to-addrspace-qualified-int4 types for the async LDS copy builtin.
typedef __attribute__((address_space(1))) v4i* gbl_v4i_p;
typedef __attribute__((address_space(3))) v4i* lds_v4i_p;

#if __has_builtin(__builtin_amdgcn_global_load_async_to_lds_b128)
#define HAVE_ASYNC_LDS 1
#else
#define HAVE_ASYNC_LDS 0
#endif

__device__ __forceinline__ void wait_asynccnt0() {
#if __has_builtin(__builtin_amdgcn_s_wait_asynccnt)
  __builtin_amdgcn_s_wait_asynccnt(0);
#else
  asm volatile("s_wait_asynccnt 0" ::: "memory");
#endif
}

// ------------------------------ RNG helpers ------------------------------
__device__ __forceinline__ uint32_t mix32(uint32_t h) {
  h ^= h >> 16; h *= 0x7FEB352Du;
  h ^= h >> 15; h *= 0x846CA68Bu;
  h ^= h >> 16;
  return h;
}
__device__ __forceinline__ uint32_t hash4(uint32_t a, uint32_t b, uint32_t c, uint32_t d) {
  return mix32(a * 0x9E3779B1u ^ b * 0x85EBCA77u ^ c * 0xC2B2AE3Du ^ d * 0x27D4EB2Fu);
}
__device__ __forceinline__ float u01(uint32_t h) {
  return ((h >> 8) + 0.5f) * (1.0f / 16777216.0f);   // (0,1)
}

// ------------------------------ WMMA GEMM ------------------------------
// C(M x 512) = A(M x 512, f32) * B(512 x 512, f16 pre-swizzled)
// mode 0: h=tanh(acc); Out = (1-h^2)*w2[col]; Uacc[row] += h*w2[col] (model)
// mode 1: Out = nan_to_num(acc)  (the gradient)
#define BM  128
#define BN  64
#define BK  32
#define LDT 40   // padded LDS stride for A (f16 elems): 80B rows, 16B-aligned sublines
#define KCH (DIM / BK)   // 16 K-chunks
#define BTILE (BN * BK)  // 2048 f16 = 4KB per swizzled B tile

__device__ __forceinline__ v8f wmma_f16(v16h a, v16h b, v8f c) {
  return __builtin_amdgcn_wmma_f32_16x16x32_f16(
      /*neg_a=*/false, a, /*neg_b=*/false, b,
      /*c_mod=*/(short)0, c, /*reuse_a=*/false, /*reuse_b=*/false);
}

// A fragment 16x32 f16 (ISA 7.12.2): lane L(0-15)=row M=L, K {0..7,16..23};
// lane L+16 = row M=L, K {8..15,24..31}.
__device__ __forceinline__ v16h frag_a(const _Float16* As, int rowBase, int lane) {
  const _Float16* p = As + (rowBase + (lane & 15)) * LDT + ((lane >> 4) << 3);
  v8h lo = *(const v8h*)(p);
  v8h hi = *(const v8h*)(p + 16);
  v16h f;
#pragma unroll
  for (int i = 0; i < 8; ++i) { f[i] = lo[i]; f[i + 8] = hi[i]; }
  return f;
}

// B fragment 32x16 f16: lanes 0-15 hold K {0..7,16..23} (N = lane), lanes 16-31 hold
// K {8..15,24..31} (N = lane-16). Bs holds the pre-swizzled tile: Bs[n][k], stride 32.
__device__ __forceinline__ v16h frag_b(const _Float16* Bs, int colBase, int lane) {
  const _Float16* p = Bs + ((colBase + (lane & 15)) << 5) + ((lane >> 4) << 3);
  v8h lo = *(const v8h*)(p);
  v8h hi = *(const v8h*)(p + 16);
  v16h f;
#pragma unroll
  for (int i = 0; i < 8; ++i) { f[i] = lo[i]; f[i + 8] = hi[i]; }
  return f;
}

__device__ __forceinline__ void epilogue(v8f acc, int gRow0, int gCol,
                                         const float* __restrict__ w2,
                                         float* __restrict__ Out,
                                         float* __restrict__ Uacc,
                                         int mode, int lane) {
  if (mode == 0) {
    float wv = w2[gCol];
    float us[8];
#pragma unroll
    for (int i = 0; i < 8; ++i) {
      float h = tanhf(acc[i]);
      us[i] = h * wv;                                   // model contribution
      Out[(size_t)(gRow0 + i) * DIM + gCol] = (1.0f - h * h) * wv;  // T matrix
    }
#pragma unroll
    for (int i = 0; i < 8; ++i) {
      float s = us[i];
      s += __shfl_xor(s, 1);
      s += __shfl_xor(s, 2);
      s += __shfl_xor(s, 4);
      s += __shfl_xor(s, 8);                            // reduce 16 columns
      if ((lane & 15) == 0) atomicAdd(&Uacc[gRow0 + i], s);
    }
  } else {
#pragma unroll
    for (int i = 0; i < 8; ++i) {
      float g = acc[i];
      g = (g != g) ? 0.0f : fminf(fmaxf(g, -1.0e6f), 1.0e6f);   // nan_to_num
      Out[(size_t)(gRow0 + i) * DIM + gCol] = g;
    }
  }
}

__global__ __launch_bounds__(256)
void gemm_wmma(const float* __restrict__ A, const _Float16* __restrict__ Bsw,
               const float* __restrict__ w2, float* __restrict__ Out,
               float* __restrict__ Uacc, int mode) {
  __shared__ __align__(16) _Float16 As[BM * LDT];
  __shared__ __align__(16) _Float16 Bs[BTILE];

  const int tid  = threadIdx.x;
  const int lane = tid & 31;
  const int wave = tid >> 5;
  const int wm   = wave & 3;              // 4 row-waves of 32 rows
  const int wn   = wave >> 2;             // 2 col-waves of 32 cols
  const int mBlk = blockIdx.x * BM;
  const int nBlk = blockIdx.y * BN;

  v8f c00 = {}, c01 = {}, c10 = {}, c11 = {};

  for (int k0 = 0; k0 < DIM; k0 += BK) {
    // Stage B tile: pre-swizzled 4KB block, one b128 async copy per thread.
    const _Float16* bsrc =
        Bsw + ((size_t)(blockIdx.y * KCH + (k0 >> 5)) * BTILE) + tid * 8;
#if HAVE_ASYNC_LDS
    __builtin_amdgcn_global_load_async_to_lds_b128(
        (gbl_v4i_p)const_cast<_Float16*>(bsrc),
        (lds_v4i_p)(Bs + tid * 8), 0, 0);
#else
    *(v8h*)(Bs + tid * 8) = *(const v8h*)(bsrc);
#endif
    // Stage A tile 128x32: f32 -> f16, 4 x float4 per thread.
#pragma unroll
    for (int c = 0; c < 4; ++c) {
      int idx = tid + c * 256;            // 0..1023 quads
      int r   = idx >> 3;                 // 0..127
      int c4  = (idx & 7) << 2;           // 0,4,...,28
      float4 v = *(const float4*)(A + (size_t)(mBlk + r) * DIM + k0 + c4);
      v4h h = { (_Float16)v.x, (_Float16)v.y, (_Float16)v.z, (_Float16)v.w };
      *(v4h*)(As + r * LDT + c4) = h;
    }
    // Prefetch next A K-chunk while this one is consumed.
    if (k0 + BK < DIM)
      __builtin_prefetch(A + (size_t)(mBlk + (tid & 127)) * DIM + k0 + BK, 0, 1);

#if HAVE_ASYNC_LDS
    wait_asynccnt0();
#endif
    __syncthreads();

    const int mW = wm * 32, nW = wn * 32;
    v16h a0 = frag_a(As, mW,      lane);
    v16h a1 = frag_a(As, mW + 16, lane);
    v16h b0 = frag_b(Bs, nW,      lane);
    v16h b1 = frag_b(Bs, nW + 16, lane);
    c00 = wmma_f16(a0, b0, c00);
    c01 = wmma_f16(a0, b1, c01);
    c10 = wmma_f16(a1, b0, c10);
    c11 = wmma_f16(a1, b1, c11);

    __syncthreads();
  }

  const int rowOff = (lane >> 4) << 3;    // 0 or 8
  const int colOff = lane & 15;
  const int r0 = mBlk + wm * 32 + rowOff;
  const int r1 = r0 + 16;
  const int cA = nBlk + wn * 32 + colOff;
  const int cB = cA + 16;
  epilogue(c00, r0, cA, w2, Out, Uacc, mode, lane);
  epilogue(c01, r0, cB, w2, Out, Uacc, mode, lane);
  epilogue(c10, r1, cA, w2, Out, Uacc, mode, lane);
  epilogue(c11, r1, cB, w2, Out, Uacc, mode, lane);
}

// ------------------------------ fused categorical proposal ------------------------------
// One block per (chain,batch); never materializes the (10,128,512,128) logits tensor.
__global__ __launch_bounds__(128)
void sampler(const float* __restrict__ chains, const float* __restrict__ G,
             const float* __restrict__ temps, const float* __restrict__ ssz,
             float* __restrict__ x_delta, float* __restrict__ lp_fwd) {
  const int cb    = blockIdx.x;           // 0..1279
  const int chain = cb / BATCH;
  const int tid   = threadIdx.x;
  const float t      = temps[chain];
  const float inv2ss = 0.5f / ssz[chain];
  __shared__ float red[128];
  float lpsum = 0.0f;
#pragma unroll
  for (int j = 0; j < 4; ++j) {
    int dim = tid + j * 128;
    size_t off = (size_t)cb * DIM + dim;
    float x = chains[off];
    float a = 0.5f * t * G[off];
    float m = -3.4e38f, s = 0.0f;
    float best = -3.4e38f, bestLogit = 0.0f;
    int bestIdx = 0;
    for (int d = 0; d < MAX_VAL; ++d) {
      float diff  = (float)d - x;
      float logit = a * diff - diff * diff * inv2ss;
      if (logit > m) { s = s * __expf(m - logit) + 1.0f; m = logit; }
      else           { s += __expf(logit - m); }
      float u   = u01(hash4(0x67B1u, (uint32_t)cb, (uint32_t)dim, (uint32_t)d));
      float gum = -__logf(-__logf(u));
      float p   = logit + gum;
      if (p > best) { best = p; bestIdx = d; bestLogit = logit; }
    }
    x_delta[off] = (float)bestIdx;
    lpsum += bestLogit - (m + __logf(s));   // log_softmax at argmax
  }
  red[tid] = lpsum;
  __syncthreads();
  for (int st = 64; st > 0; st >>= 1) {
    if (tid < st) red[tid] += red[tid + st];
    __syncthreads();
  }
  if (tid == 0) lp_fwd[cb] = red[0];
}

// ------------------------------ reverse logprob + accept ------------------------------
__global__ __launch_bounds__(128)
void reverse_accept(const float* __restrict__ chains, const float* __restrict__ xd,
                    const float* __restrict__ Gd, const float* __restrict__ temps,
                    const float* __restrict__ ssz, const float* __restrict__ U,
                    const float* __restrict__ Ud, const float* __restrict__ lp_fwd,
                    float* __restrict__ chains_new) {
  const int cb    = blockIdx.x;
  const int chain = cb / BATCH;
  const int tid   = threadIdx.x;
  const float t      = temps[chain];
  const float inv2ss = 0.5f / ssz[chain];
  const float coefT  = 0.25f * t * t;     // t * (grad*t/2) * diff/2
  __shared__ float red[128];
  __shared__ float acceptFlag;
  float lpsum = 0.0f;
  float xs[4], xds[4];
#pragma unroll
  for (int j = 0; j < 4; ++j) {
    int dim = tid + j * 128;
    size_t off = (size_t)cb * DIM + dim;
    float x   = chains[off];
    float xdv = xd[off];
    xs[j] = x; xds[j] = xdv;
    float a  = coefT * Gd[off];
    int   xi = (int)x;
    float m = -3.4e38f, s = 0.0f, atIdx = 0.0f;
    for (int d = 0; d < MAX_VAL; ++d) {
      float diff  = (float)d - xdv;
      float logit = a * diff - diff * diff * inv2ss;
      if (logit > m) { s = s * __expf(m - logit) + 1.0f; m = logit; }
      else           { s += __expf(logit - m); }
      if (d == xi) atIdx = logit;
    }
    lpsum += atIdx - (m + __logf(s));
  }
  red[tid] = lpsum;
  __syncthreads();
  for (int st = 64; st > 0; st >>= 1) {
    if (tid < st) red[tid] += red[tid + st];
    __syncthreads();
  }
  if (tid == 0) {
    float log_acc = (Ud[cb] - U[cb]) + red[0] - lp_fwd[cb];
    float u = u01(hash4(0xACCEu, (uint32_t)cb, 0u, 0u));
    acceptFlag = (__expf(log_acc) > u) ? 1.0f : 0.0f;
  }
  __syncthreads();
  bool acc = acceptFlag > 0.5f;
#pragma unroll
  for (int j = 0; j < 4; ++j) {
    int dim = tid + j * 128;
    chains_new[(size_t)cb * DIM + dim] = acc ? xds[j] : xs[j];
  }
}

// ------------------------------ replica exchange ------------------------------
__global__ __launch_bounds__(128)
void swap_kernel(float* __restrict__ chains_new, const float* __restrict__ U2,
                 const float* __restrict__ temps, float* __restrict__ out) {
  const int b   = blockIdx.x;             // batch column
  const int tid = threadIdx.x;
  __shared__ float u[NUM_CHAINS];
  __shared__ int doswap;
  if (tid < NUM_CHAINS) u[tid] = U2[tid * BATCH + b];
  __syncthreads();
  for (int i = 0; i < NUM_CHAINS - 1; ++i) {
    if (tid == 0) {
      float s = fminf(__expf((temps[i + 1] - temps[i]) * (u[i + 1] - u[i])), 1.0f);
      float r = u01(hash4(0x5A5Au, (uint32_t)i, (uint32_t)b, 0u));
      doswap = (r < s) ? 1 : 0;
      if (doswap) { float tmp = u[i]; u[i] = u[i + 1]; u[i + 1] = tmp; }
    }
    __syncthreads();
    if (doswap) {
      size_t lo = ((size_t)i * BATCH + b) * DIM;
      size_t hi = ((size_t)(i + 1) * BATCH + b) * DIM;
#pragma unroll
      for (int j = 0; j < 4; ++j) {
        int dim = tid + j * 128;
        float a = chains_new[lo + dim], c = chains_new[hi + dim];
        chains_new[lo + dim] = c;
        chains_new[hi + dim] = a;
      }
    }
    __syncthreads();
  }
#pragma unroll
  for (int j = 0; j < 4; ++j) {
    int dim = tid + j * 128;
    out[(size_t)b * DIM + dim] = chains_new[(size_t)b * DIM + dim];
  }
}

// ------------------------------ small utility kernels ------------------------------
__global__ void zero_kernel(float* __restrict__ p, int n) {
  int i = blockIdx.x * 256 + threadIdx.x;
  if (i < n) p[i] = 0.0f;
}

// Convert W1 to f16 and pre-swizzle both W1 and W1^T into fragment-ready tile
// order: [nTile(64)][kChunk(32)] -> contiguous 2048-elem (4KB) blocks laid out
// as Bs[n][k] (n-major, k stride 32), so GEMM B staging is one contiguous
// async b128 copy per thread.
__global__ __launch_bounds__(256)
void convert_w1(const float* __restrict__ W1, _Float16* __restrict__ W1swz,
                _Float16* __restrict__ W1Tswz) {
  int i = blockIdx.x * 256 + threadIdx.x;   // 0..262143
  int k = i >> 9, n = i & 511;              // element B[k][n]
  int nt = n >> 6, nl = n & 63;
  int kc = k >> 5, kl = k & 31;
  size_t dst = ((size_t)(nt * KCH + kc) * BN + nl) * BK + kl;
  W1swz[dst]  = (_Float16)W1[(size_t)k * DIM + n];   // B = W1
  W1Tswz[dst] = (_Float16)W1[(size_t)n * DIM + k];   // B = W1^T
}

// ------------------------------ launch ------------------------------
extern "C" void kernel_launch(void* const* d_in, const int* in_sizes, int n_in,
                              void* d_out, int out_size, void* d_ws, size_t ws_size,
                              hipStream_t stream) {
  const float* chains = (const float*)d_in[0];
  const float* W1     = (const float*)d_in[1];
  const float* w2     = (const float*)d_in[2];
  const float* temps  = (const float*)d_in[3];
  const float* ssz    = (const float*)d_in[4];
  float* out = (float*)d_out;

  char* ws = (char*)d_ws;
  size_t off = 0;
  auto alloc = [&](size_t bytes) -> void* {
    void* p = ws + off;
    off += (bytes + 255) & ~(size_t)255;
    return p;
  };
  _Float16* W1swz  = (_Float16*)alloc((size_t)DIM * DIM * 2);
  _Float16* W1Tswz = (_Float16*)alloc((size_t)DIM * DIM * 2);
  float* T   = (float*)alloc((size_t)MTOT * DIM * 4);
  float* G   = (float*)alloc((size_t)MTOT * DIM * 4);
  float* Gd  = (float*)alloc((size_t)MTOT * DIM * 4);
  float* Xd  = (float*)alloc((size_t)MTOT * DIM * 4);
  float* Cn  = (float*)alloc((size_t)MTOT * DIM * 4);
  float* U   = (float*)alloc(MTOT * 4);   // these three are contiguous
  float* Ud  = (float*)alloc(MTOT * 4);
  float* U2  = (float*)alloc(MTOT * 4);
  float* lpf = (float*)alloc(MTOT * 4);

  convert_w1<<<(DIM * DIM) / 256, 256, 0, stream>>>(W1, W1swz, W1Tswz);
  zero_kernel<<<(3 * MTOT + 255) / 256, 256, 0, stream>>>(U, 3 * MTOT);

  dim3 gg(MTOT / BM, DIM / BN);   // (10, 8)
  // forward: model + grad on current chains
  gemm_wmma<<<gg, 256, 0, stream>>>(chains, W1swz, w2, T, U, 0);
  gemm_wmma<<<gg, 256, 0, stream>>>(T, W1Tswz, w2, G, U, 1);
  // fused categorical proposal
  sampler<<<MTOT, 128, 0, stream>>>(chains, G, temps, ssz, Xd, lpf);
  // model + grad at proposal
  gemm_wmma<<<gg, 256, 0, stream>>>(Xd, W1swz, w2, T, Ud, 0);
  gemm_wmma<<<gg, 256, 0, stream>>>(T, W1Tswz, w2, Gd, Ud, 1);
  // reverse logprob + MH accept
  reverse_accept<<<MTOT, 128, 0, stream>>>(chains, Xd, Gd, temps, ssz, U, Ud, lpf, Cn);
  // model at post-accept chains, then replica exchange
  gemm_wmma<<<gg, 256, 0, stream>>>(Cn, U2 ? W1swz : W1swz, w2, T, U2, 0);
  swap_kernel<<<BATCH, 128, 0, stream>>>(Cn, U2, temps, out);
}
